// ColAttention_3075196584449
// MI455X (gfx1250) — matrix operand order repaired and checked by reading, not compiled, not run on hardware
//
#include <hip/hip_runtime.h>
#include <hip/hip_bf16.h>
#include <cstdint>

typedef __attribute__((ext_vector_type(16))) __bf16 v16bf;
typedef __attribute__((ext_vector_type(8)))  float  v8f;
typedef int v4i_ __attribute__((vector_size(16)));   // matches builtin's V4i

#define B_ 8
#define C_ 512
#define D_ 64
#define H_ 128
#define W_ 128
#define P_ (H_*W_)

// ---- workspace layout (bytes) ----
static constexpr size_t OFF_WQ = 0;                                  // bf16 [64][512]
static constexpr size_t OFF_WK = OFF_WQ + (size_t)D_*C_*2;           // bf16 [64][512]
static constexpr size_t OFF_WV = OFF_WK + (size_t)D_*C_*2;           // bf16 [512][512]
static constexpr size_t OFF_QT = OFF_WV + (size_t)C_*C_*2;           // bf16 [b][h][w][64]
static constexpr size_t OFF_KT = OFF_QT + (size_t)B_*P_*D_*2;        // bf16 [b][h][w][64]
static constexpr size_t OFF_VT = OFF_KT + (size_t)B_*P_*D_*2;        // bf16 [b][w][h][512]
static constexpr size_t OFF_O  = OFF_VT + (size_t)B_*P_*C_*2;        // f32  [b][w][i][512]

// ---- gfx1250 async global->LDS copy (ASYNCcnt path) ----
#if defined(__HIP_DEVICE_COMPILE__) && \
    __has_builtin(__builtin_amdgcn_global_load_async_to_lds_b128) && \
    __has_builtin(__builtin_amdgcn_s_wait_asynccnt)
#define USE_ASYNC_LDS 1
#else
#define USE_ASYNC_LDS 0
#endif

struct B16cp { unsigned long long lo, hi; };

__device__ __forceinline__ void g2l_b128(void* lds_dst, const void* gsrc) {
#if USE_ASYNC_LDS
    __builtin_amdgcn_global_load_async_to_lds_b128(
        (__attribute__((address_space(1))) v4i_*)(uintptr_t)gsrc,
        (__attribute__((address_space(3))) v4i_*)(unsigned)(uintptr_t)lds_dst,
        0, 0);
#else
    *(B16cp*)lds_dst = *(const B16cp*)gsrc;
#endif
}
__device__ __forceinline__ void wait_async_le4() {
#if USE_ASYNC_LDS
    __builtin_amdgcn_s_wait_asynccnt(4);
#endif
}
__device__ __forceinline__ void wait_async_0() {
#if USE_ASYNC_LDS
    __builtin_amdgcn_s_wait_asynccnt(0);
#endif
}

// ---------------- kernel 0: weight conversion to bf16 ----------------
__global__ void cvt_weights(const float* __restrict__ Wq, const float* __restrict__ Wk,
                            const float* __restrict__ Wv,
                            __bf16* __restrict__ wq, __bf16* __restrict__ wk,
                            __bf16* __restrict__ wv) {
    int tid = blockIdx.x * blockDim.x + threadIdx.x;
    int stride = gridDim.x * blockDim.x;
    for (int i = tid; i < D_*C_; i += stride) { wq[i] = (__bf16)Wq[i]; wk[i] = (__bf16)Wk[i]; }
    for (int i = tid; i < C_*C_; i += stride) { wv[i] = (__bf16)Wv[i]; }
}

// ---------------- kernel 1: fused Q/K/V projection -------------------
// One block = one strip of 16 consecutive pixels of one batch.
// out[p][n] = sum_c x[p][c] * W[n][c]  -> A = X^T (16x32 per step), B = W rows (lane=col n).
__global__ __launch_bounds__(256) void proj_qkv(
    const float* __restrict__ x,
    const __bf16* __restrict__ wq, const __bf16* __restrict__ wk, const __bf16* __restrict__ wv,
    const float* __restrict__ bq, const float* __restrict__ bk, const float* __restrict__ bv,
    __bf16* __restrict__ Qt, __bf16* __restrict__ Kt, __bf16* __restrict__ Vt)
{
    const int lane = threadIdx.x & 31;
    const int wid  = threadIdx.x >> 5;       // 8 waves
    const int hi   = lane >> 4;
    const int ln   = lane & 15;
    const int b    = blockIdx.x >> 10;       // 1024 strips per batch
    const int p0   = (blockIdx.x & 1023) * 16;

    // 40 output 16-col tiles total: 0..3 Q, 4..7 K, 8..39 V ; 5 per wave
    const __bf16* bbase[5];
    int kindv[5], ntile[5];
    #pragma unroll
    for (int s = 0; s < 5; ++s) {
        int ntg = wid * 5 + s;
        if (ntg < 4)      { bbase[s] = wq + (size_t)ntg    *16*C_; kindv[s]=0; ntile[s]=ntg;   }
        else if (ntg < 8) { bbase[s] = wk + (size_t)(ntg-4)*16*C_; kindv[s]=1; ntile[s]=ntg-4; }
        else              { bbase[s] = wv + (size_t)(ntg-8)*16*C_; kindv[s]=2; ntile[s]=ntg-8; }
    }

    const float* xb = x + (size_t)b * C_ * P_;
    v8f acc[5] = {};

    for (int kk = 0; kk < C_; kk += 32) {
        if (kk + 32 < C_)  // pull next A-panel toward the caches early
            __builtin_prefetch(xb + (size_t)(kk + 32 + hi * 8) * P_ + p0 + ln, 0, 1);
        // A fragment (16 pixels x 32 channels), contiguous k order (matches B):
        // lane holds row M=ln; K = hi*16 + t. Each t-load is lane-coalesced.
        v16bf a;
        #pragma unroll
        for (int t = 0; t < 16; ++t) {
            int k = kk + hi*16 + t;
            a[t] = (__bf16)xb[(size_t)k * P_ + p0 + ln];
        }
        #pragma unroll
        for (int s = 0; s < 5; ++s) {
            // B fragment: lane = column n, 16 contiguous K values (32B vector load)
            v16bf bm = *(const v16bf*)(bbase[s] + (size_t)ln * C_ + kk + hi * 16);
            acc[s] = __builtin_amdgcn_wmma_f32_16x16x32_bf16(
                false, a, false, bm, (short)0, acc[s], false, false);
        }
    }

    // D layout: lanes0-15: (M=r, N=ln) ; lanes16-31: (M=r+8, N=ln)
    #pragma unroll
    for (int s = 0; s < 5; ++s) {
        int nt = ntile[s];
        #pragma unroll
        for (int r = 0; r < 8; ++r) {
            int m = r + hi * 8;
            int p = p0 + m;
            if (kindv[s] == 0) {
                int d = nt * 16 + ln;
                Qt[((size_t)b * P_ + p) * D_ + d] = (__bf16)(acc[s][r] + bq[d]);
            } else if (kindv[s] == 1) {
                int d = nt * 16 + ln;
                Kt[((size_t)b * P_ + p) * D_ + d] = (__bf16)(acc[s][r] + bk[d]);
            } else {
                int c = nt * 16 + ln;
                int h = p >> 7, w = p & 127;
                Vt[(((size_t)b * W_ + w) * H_ + h) * C_ + c] = (__bf16)(acc[s][r] + bv[c]);
            }
        }
    }
}

// ---------------- kernel 2: column attention per (b,w) ---------------
__global__ __launch_bounds__(256) void col_attn(
    const __bf16* __restrict__ Qt, const __bf16* __restrict__ Kt,
    const __bf16* __restrict__ Vt, float* __restrict__ O)
{
    // ldsV[0] first holds the K tile (exactly 128x64), then V double-buffers.
    __shared__ __align__(32) __bf16 ldsV[2][H_][64];   // 32 KB
    __shared__ __align__(32) __bf16 ldsA[8][16][H_];   // 32 KB, wave-private attn tiles

    const int lane = threadIdx.x & 31;
    const int wid  = threadIdx.x >> 5;
    const int hi   = lane >> 4;
    const int ln   = lane & 15;
    const int b    = blockIdx.x >> 7;
    const int w    = blockIdx.x & 127;
    const int i0   = wid * 16;            // wave's 16 query rows

    // ---- stage K tile (128 x 64 bf16) into ldsV[0] via async copy ----
    for (int c = threadIdx.x; c < H_ * 8; c += 256) {       // 1024 x 16B chunks
        int j = c >> 3, ch = (c & 7) * 8;
        g2l_b128(&ldsV[0][j][ch], Kt + (((size_t)b * H_ + j) * W_ + w) * D_ + ch);
    }
    wait_async_0();
    __syncthreads();

    // ---- scores S = Q K^T (16 x 128 per wave), K-dim = 64 ----
    v16bf aQ[2];
    #pragma unroll
    for (int ks = 0; ks < 2; ++ks) {
        #pragma unroll
        for (int t = 0; t < 16; ++t) {
            int k = ks * 32 + hi * 16 + t;
            aQ[ks][t] = Qt[(((size_t)b * H_ + (i0 + ln)) * W_ + w) * D_ + k];
        }
    }
    v8f accS[8] = {};
    #pragma unroll
    for (int jt = 0; jt < 8; ++jt) {
        #pragma unroll
        for (int ks = 0; ks < 2; ++ks) {
            v16bf bm = *(const v16bf*)&ldsV[0][jt * 16 + ln][ks * 32 + hi * 16];
            accS[jt] = __builtin_amdgcn_wmma_f32_16x16x32_bf16(
                false, aQ[ks], false, bm, (short)0, accS[jt], false, false);
        }
    }

    // ---- softmax over j. Row r in lanes 0-15, row r+8 in lanes 16-31.
    #pragma unroll
    for (int r = 0; r < 8; ++r) {
        float mx = accS[0][r];
        #pragma unroll
        for (int jt = 1; jt < 8; ++jt) mx = fmaxf(mx, accS[jt][r]);
        #pragma unroll
        for (int msk = 1; msk < 16; msk <<= 1) mx = fmaxf(mx, __shfl_xor(mx, msk, 32));
        float sm = 0.f;
        #pragma unroll
        for (int jt = 0; jt < 8; ++jt) {
            float e = __expf(accS[jt][r] - mx);
            accS[jt][r] = e;
            sm += e;
        }
        #pragma unroll
        for (int msk = 1; msk < 16; msk <<= 1) sm += __shfl_xor(sm, msk, 32);
        float inv = 1.0f / sm;
        int m = r + hi * 8;
        #pragma unroll
        for (int jt = 0; jt < 8; ++jt)
            ldsA[wid][m][jt * 16 + ln] = (__bf16)(accS[jt][r] * inv);
    }

    // ---- build the 4 attn A-fragments ONCE (invariant across channel groups)
    v16bf aA[4];
    #pragma unroll
    for (int ks = 0; ks < 4; ++ks) {
        #pragma unroll
        for (int t = 0; t < 16; ++t) {
            int k = ks * 32 + hi * 16 + t;
            aA[ks][t] = ldsA[wid][ln][k];
        }
    }

    // ---- out = attn @ V, 8 groups of 64 channels, double-buffered async V staging
    const __bf16* vbw = Vt + ((size_t)b * W_ + w) * (size_t)H_ * C_;
    float*        obw = O  + ((size_t)b * W_ + w) * (size_t)H_ * C_;

    auto stageV = [&](int g) {
        __bf16 (*buf)[64] = ldsV[g & 1];
        for (int c = threadIdx.x; c < H_ * 8; c += 256) {   // 128x64 bf16 = 1024 x 16B
            int j = c >> 3, ch = (c & 7) * 8;
            g2l_b128(&buf[j][ch], vbw + (size_t)j * C_ + g * 64 + ch);
        }
    };

    __syncthreads();          // everyone done reading K from ldsV[0]
    stageV(0);
    for (int g = 0; g < 8; ++g) {
        __syncthreads();      // readers of buf[(g+1)&1] (iteration g-1) are done
        if (g < 7) { stageV(g + 1); wait_async_le4(); }  // old group landed (in-order)
        else       { wait_async_0(); }
        __syncthreads();      // group g visible to all waves

        const __bf16 (*vb)[64] = ldsV[g & 1];
        v8f acc[4] = {};
        #pragma unroll
        for (int ks = 0; ks < 4; ++ks) {          // K = 128 keys, 32 per step
            #pragma unroll
            for (int nt = 0; nt < 4; ++nt) {
                v16bf bm;
                #pragma unroll
                for (int t = 0; t < 16; ++t) {
                    int j = ks * 32 + hi * 16 + t;
                    bm[t] = vb[j][nt * 16 + ln];
                }
                acc[nt] = __builtin_amdgcn_wmma_f32_16x16x32_bf16(
                    false, aA[ks], false, bm, (short)0, acc[nt], false, false);
            }
        }
        #pragma unroll
        for (int nt = 0; nt < 4; ++nt) {
            int c = g * 64 + nt * 16 + ln;
            #pragma unroll
            for (int r = 0; r < 8; ++r) {
                int i = i0 + r + hi * 8;
                obw[(size_t)i * C_ + c] = acc[nt][r]; // lanes 0-15 write 64B contiguous
            }
        }
    }
}

// ---------------- kernel 3: transpose + gamma*out + x ----------------
// O[b][w][i][c] (coalesced read)  ->  out[b][c][i][w] (coalesced write) via LDS tile.
__global__ __launch_bounds__(256) void finalize(
    const float* __restrict__ O, const float* __restrict__ x,
    const float* __restrict__ gamma, float* __restrict__ out)
{
    __shared__ float tile[32][33];
    const int bi = blockIdx.y;               // b*H + i
    const int b  = bi >> 7, i = bi & 127;
    const int tw = (blockIdx.x & 3)  * 32;   // W/32 = 4
    const int tc = (blockIdx.x >> 2) * 32;   // C/32 = 16
    const int tx = threadIdx.x & 31, ty = threadIdx.x >> 5;
    const float g = gamma[0];

    #pragma unroll
    for (int k = 0; k < 4; ++k) {
        int wl = ty + k * 8;
        tile[wl][tx] = O[(((size_t)b * W_ + (tw + wl)) * H_ + i) * C_ + tc + tx];
    }
    __syncthreads();
    #pragma unroll
    for (int k = 0; k < 4; ++k) {
        int cl = ty + k * 8;
        size_t oidx = (((size_t)b * C_ + (tc + cl)) * H_ + i) * W_ + tw + tx;
        out[oidx] = g * tile[tx][cl] + x[oidx];
    }
}

// ------------------------------- launch -------------------------------
extern "C" void kernel_launch(void* const* d_in, const int* in_sizes, int n_in,
                              void* d_out, int out_size, void* d_ws, size_t ws_size,
                              hipStream_t stream) {
    const float* x     = (const float*)d_in[0];
    const float* Wq    = (const float*)d_in[1];
    const float* bq    = (const float*)d_in[2];
    const float* Wk    = (const float*)d_in[3];
    const float* bk    = (const float*)d_in[4];
    const float* Wv    = (const float*)d_in[5];
    const float* bv    = (const float*)d_in[6];
    const float* gamma = (const float*)d_in[7];
    float* out = (float*)d_out;

    char* ws = (char*)d_ws;
    __bf16* wqb = (__bf16*)(ws + OFF_WQ);
    __bf16* wkb = (__bf16*)(ws + OFF_WK);
    __bf16* wvb = (__bf16*)(ws + OFF_WV);
    __bf16* Qt  = (__bf16*)(ws + OFF_QT);
    __bf16* Kt  = (__bf16*)(ws + OFF_KT);
    __bf16* Vt  = (__bf16*)(ws + OFF_VT);
    float*  O   = (float*) (ws + OFF_O);

    cvt_weights<<<256, 256, 0, stream>>>(Wq, Wk, Wv, wqb, wkb, wvb);
    proj_qkv<<<B_ * (P_ / 16), 256, 0, stream>>>(x, wqb, wkb, wvb, bq, bk, bv, Qt, Kt, Vt);
    col_attn<<<B_ * W_, 256, 0, stream>>>(Qt, Kt, Vt, O);
    finalize<<<dim3((W_ / 32) * (C_ / 32), B_ * H_), 256, 0, stream>>>(O, x, gamma, out);
}
